// my_model_88416196756160
// MI455X (gfx1250) — compile-verified, matrix operand order checked by reading, not run on hardware
//
#include <hip/hip_runtime.h>
#include <hip/hip_bf16.h>

// T=192, B=16, IN=256, H=256, 4H=1024, IN+H=512, IN+2H=768
#define TT 192
#define BB 16
#define HH 256

typedef _Float16 h16;
typedef __attribute__((ext_vector_type(16))) _Float16 v16h;
typedef __attribute__((ext_vector_type(8)))  float    v8f;

__device__ __forceinline__ v8f wmma16(v16h a, v16h b, v8f c) {
  return __builtin_amdgcn_wmma_f32_16x16x32_f16(false, a, false, b, (short)0, c, false, false);
}

// Build an A-fragment (16xK tile slice, M=batch rows) from f32 activations in LDS.
// Layout per ISA 7.12.2 (16-bit A 16x32): lane L holds row m=L&15;
// element j -> k = kbase + (j>>3)*16 + (j&7) + 8*(L>>4).
__device__ __forceinline__ v16h load_a_frag(const float* act, int ldk, int kbase, int lane) {
  int m = lane & 15;
  int g = lane >> 4;
  const float* p = act + m * ldk + kbase + 8 * g;
  float4 q0 = *(const float4*)(p);
  float4 q1 = *(const float4*)(p + 4);
  float4 q2 = *(const float4*)(p + 16);
  float4 q3 = *(const float4*)(p + 20);
  v16h a;
  a[0]=(h16)q0.x; a[1]=(h16)q0.y; a[2]=(h16)q0.z; a[3]=(h16)q0.w;
  a[4]=(h16)q1.x; a[5]=(h16)q1.y; a[6]=(h16)q1.z; a[7]=(h16)q1.w;
  a[8]=(h16)q2.x; a[9]=(h16)q2.y; a[10]=(h16)q2.z; a[11]=(h16)q2.w;
  a[12]=(h16)q3.x; a[13]=(h16)q3.y; a[14]=(h16)q3.z; a[15]=(h16)q3.w;
  return a;
}

// Pre-packed B fragment: contiguous 16 halfs per lane per tile (two b128 loads).
__device__ __forceinline__ v16h load_b_frag(const h16* pack, int tile, int lane) {
  return *(const v16h*)(pack + (size_t)tile * 512 + lane * 16);
}

__device__ __forceinline__ float sigm(float v) { return 1.0f / (1.0f + __expf(-v)); }

// -------- weight packing: W[N][K] row-major f32 -> B fragments [nt][kt][lane][16] f16.
// Column k < K_A comes from srcA (width K_A), else srcB (width K-K_A).
__global__ void pack_weights(h16* __restrict__ dst, const float* __restrict__ srcA,
                             const float* __restrict__ srcB,
                             int K_A, int K, int NT, int KT) {
  int tid = blockIdx.x * blockDim.x + threadIdx.x;
  int total = NT * KT * 512;
  if (tid >= total) return;
  int j = tid & 15;
  int lane = (tid >> 4) & 31;
  int tile = tid >> 9;
  int nt = tile / KT;
  int kt = tile - nt * KT;
  int g = lane >> 4;
  int k = kt * 32 + ((j >> 3) * 16 + (j & 7) + 8 * g);
  int n = nt * 16 + (lane & 15);
  float v = (k < K_A) ? srcA[(size_t)n * K_A + k]
                      : srcB[(size_t)n * (K - K_A) + (k - K_A)];
  dst[tid] = (h16)v;
}

// -------- shared LSTM scan: one persistent workgroup, 16 waves.
// Wave w owns hidden columns [16w,16w+16); its 4 N-tiles are the i/f/g/o gate
// blocks for those columns -> gate fusion fully in-register; c stays in a v8f.
__global__ void __launch_bounds__(512) shared_lstm_kernel(
    const float* __restrict__ x, const h16* __restrict__ W1p,
    const float* __restrict__ b_lstm, float* __restrict__ hs) {
  __shared__ __align__(16) float act[16 * 512];   // [m][ x(256) | h(256) ]
  int tid = threadIdx.x;
  int lane = tid & 31;
  int w = tid >> 5;
  int ncol = w * 16 + (lane & 15);
  float bi = b_lstm[ncol], bf = b_lstm[256 + ncol];
  float bg = b_lstm[512 + ncol], bo = b_lstm[768 + ncol];
  for (int i = tid; i < 16 * 256; i += 512) act[(i >> 8) * 512 + 256 + (i & 255)] = 0.0f;
  v8f cfrag = {0.f,0.f,0.f,0.f,0.f,0.f,0.f,0.f};
  for (int t = 0; t < TT; ++t) {
    const float4* xv = (const float4*)(x + (size_t)t * 4096);
    for (int i = tid; i < 1024; i += 512) {
      int m = i >> 6, kq = i & 63;
      *(float4*)(act + m * 512 + kq * 4) = xv[i];
    }
    __syncthreads();
    v8f a0 = {0.f,0.f,0.f,0.f,0.f,0.f,0.f,0.f}, a1 = a0, a2 = a0, a3 = a0;
    for (int kt = 0; kt < 16; ++kt) {
      v16h a = load_a_frag(act, 512, kt * 32, lane);
      a0 = wmma16(a, load_b_frag(W1p, (w)      * 16 + kt, lane), a0);
      a1 = wmma16(a, load_b_frag(W1p, (w + 16) * 16 + kt, lane), a1);
      a2 = wmma16(a, load_b_frag(W1p, (w + 32) * 16 + kt, lane), a2);
      a3 = wmma16(a, load_b_frag(W1p, (w + 48) * 16 + kt, lane), a3);
    }
    __syncthreads();  // all waves done reading act before h update
#pragma unroll
    for (int r = 0; r < 8; ++r) {
      int m = r + ((lane >> 4) << 3);
      float iv = sigm(a0[r] + bi);
      float fv = sigm(a1[r] + bf);
      float gv = tanhf(a2[r] + bg);
      float ov = sigm(a3[r] + bo);
      float cn = fv * cfrag[r] + iv * gv;
      cfrag[r] = cn;
      float hn = ov * tanhf(cn);
      act[m * 512 + 256 + ncol] = hn;
      hs[((size_t)t * 16 + m) * 256 + ncol] = hn;
    }
    __syncthreads();
  }
}

// -------- step-invariant attention projection: base[t] = hs[t]*Ws1^T + x[t]*Ws3^T + Ws_b
// Fully parallel across 192 workgroups (fills the chip).
__global__ void __launch_bounds__(256) base_kernel(
    const float* __restrict__ hs, const float* __restrict__ x,
    const h16* __restrict__ Wsp, const float* __restrict__ Ws_b,
    float* __restrict__ base) {
  __shared__ __align__(16) float act[16 * 512];   // [m][ hs(256) | x(256) ]
  int t = blockIdx.x;
  int tid = threadIdx.x;
  int lane = tid & 31;
  int w = tid >> 5;     // 8 waves
  const float4* hv = (const float4*)(hs + (size_t)t * 4096);
  const float4* xv = (const float4*)(x  + (size_t)t * 4096);
  for (int i = tid; i < 1024; i += 256) {
    int m = i >> 6, kq = i & 63;
    *(float4*)(act + m * 512 + kq * 4)       = hv[i];
    *(float4*)(act + m * 512 + 256 + kq * 4) = xv[i];
  }
  __syncthreads();
  for (int half = 0; half < 2; ++half) {
    int nt = w + 8 * half;
    v8f acc = {0.f,0.f,0.f,0.f,0.f,0.f,0.f,0.f};
    for (int kt = 0; kt < 8; ++kt) {          // hs part: Ws columns [0,H)
      v16h a = load_a_frag(act, 512, kt * 32, lane);
      acc = wmma16(a, load_b_frag(Wsp, nt * 24 + kt, lane), acc);
    }
    for (int kt = 0; kt < 8; ++kt) {          // x part: Ws columns [2H,2H+IN)
      v16h a = load_a_frag(act, 512, 256 + kt * 32, lane);
      acc = wmma16(a, load_b_frag(Wsp, nt * 24 + 16 + kt, lane), acc);
    }
    int n = nt * 16 + (lane & 15);
    float bias = Ws_b[n];
#pragma unroll
    for (int r = 0; r < 8; ++r) {
      int m = r + ((lane >> 4) << 3);
      base[((size_t)t * 16 + m) * 256 + n] = acc[r] + bias;
    }
  }
}

// -------- task scan: one persistent workgroup. Per step:
// s2 = h*Ws2^T (WMMA) -> scores via tanh(base+s2)@Us -> softmax(T) -> Rt -> message LSTM (WMMA).
__global__ void __launch_bounds__(512) task_kernel(
    const float* __restrict__ x, const float* __restrict__ hs,
    const float* __restrict__ base, const h16* __restrict__ Wsp,
    const float* __restrict__ Us_w, const float* __restrict__ Us_b,
    const h16* __restrict__ Wcp, const float* __restrict__ bc,
    float* __restrict__ out) {
  __shared__ __align__(16) float act[16 * 768];   // [m][ x(256) | s2/Rt(256) | h(256) ]
  __shared__ float score[TT * 16];
  __shared__ float us[256];
  __shared__ float colmax[16], colinv[16];
  int tid = threadIdx.x;
  int lane = tid & 31;
  int w = tid >> 5;      // 16 waves
  int ncol = w * 16 + (lane & 15);
  float bi = bc[ncol], bf = bc[256 + ncol], bg = bc[512 + ncol], bo = bc[768 + ncol];
  float usb = Us_b[0];
  for (int i = tid; i < 256; i += 512) us[i] = Us_w[i];
  for (int i = tid; i < 16 * 256; i += 512) act[(i >> 8) * 768 + 512 + (i & 255)] = 0.0f;
  v8f cfrag = {0.f,0.f,0.f,0.f,0.f,0.f,0.f,0.f};
  __syncthreads();
  for (int t = 0; t < TT; ++t) {
    // 1) s2 = h @ Ws2^T  (Ws columns [H,2H) -> kt 8..15); wave w owns N-tile w
    v8f accs = {0.f,0.f,0.f,0.f,0.f,0.f,0.f,0.f};
    for (int kt = 0; kt < 8; ++kt) {
      v16h a = load_a_frag(act, 768, 512 + kt * 32, lane);
      accs = wmma16(a, load_b_frag(Wsp, w * 24 + 8 + kt, lane), accs);
    }
#pragma unroll
    for (int r = 0; r < 8; ++r) {
      int m = r + ((lane >> 4) << 3);
      act[m * 768 + 256 + ncol] = accs[r];     // s2 parked in Rt slot
    }
    const float4* xv = (const float4*)(x + (size_t)t * 4096);
    for (int i = tid; i < 1024; i += 512) {
      int m = i >> 6, kq = i & 63;
      *(float4*)(act + m * 768 + kq * 4) = xv[i];
    }
    __syncthreads();
    // 2) scores[t'][b] = tanh(base[t',b,:]+s2[b,:]) . us + usb   (3072 pairs, 6/thread)
    for (int p = tid; p < TT * 16; p += 512) {
      int tp = p >> 4, b = p & 15;
      const float* bs = base + ((size_t)tp * 16 + b) * 256;
      const float* s2 = act + b * 768 + 256;
      float s = usb;
      for (int hh = 0; hh < 256; ++hh) s += tanhf(bs[hh] + s2[hh]) * us[hh];
      score[p] = s;
    }
    __syncthreads();
    // 3) softmax over time per batch column
    if (tid < 16) {
      float mx = -1e30f;
      for (int tp = 0; tp < TT; ++tp) mx = fmaxf(mx, score[tp * 16 + tid]);
      colmax[tid] = mx;
    }
    __syncthreads();
    for (int p = tid; p < TT * 16; p += 512) score[p] = __expf(score[p] - colmax[p & 15]);
    __syncthreads();
    if (tid < 16) {
      float sm = 0.f;
      for (int tp = 0; tp < TT; ++tp) sm += score[tp * 16 + tid];
      colinv[tid] = 1.0f / sm;
    }
    __syncthreads();
    // 4) Rt[b][h] = (1/sum) * sum_t' e[t'][b]*hs[t'][b][h]  (overwrites dead s2)
    for (int p = tid; p < 4096; p += 512) {
      int b = p >> 8, hh = p & 255;
      const float* hp = hs + b * 256 + hh;
      float acc = 0.f;
      for (int tp = 0; tp < TT; ++tp) acc += score[tp * 16 + b] * hp[(size_t)tp * 4096];
      act[b * 768 + 256 + hh] = acc * colinv[b];
    }
    __syncthreads();
    // 5) message LSTM: [x_t,Rt,h] (K=768) @ Wc^T ; wave w fuses its i/f/g/o tiles
    v8f a0 = {0.f,0.f,0.f,0.f,0.f,0.f,0.f,0.f}, a1 = a0, a2 = a0, a3 = a0;
    for (int kt = 0; kt < 24; ++kt) {
      v16h a = load_a_frag(act, 768, kt * 32, lane);
      a0 = wmma16(a, load_b_frag(Wcp, (w)      * 24 + kt, lane), a0);
      a1 = wmma16(a, load_b_frag(Wcp, (w + 16) * 24 + kt, lane), a1);
      a2 = wmma16(a, load_b_frag(Wcp, (w + 32) * 24 + kt, lane), a2);
      a3 = wmma16(a, load_b_frag(Wcp, (w + 48) * 24 + kt, lane), a3);
    }
    __syncthreads();
#pragma unroll
    for (int r = 0; r < 8; ++r) {
      int m = r + ((lane >> 4) << 3);
      float iv = sigm(a0[r] + bi);
      float fv = sigm(a1[r] + bf);
      float gv = tanhf(a2[r] + bg);
      float ov = sigm(a3[r] + bo);
      float cn = fv * cfrag[r] + iv * gv;
      cfrag[r] = cn;
      float hn = ov * tanhf(cn);
      act[m * 768 + 512 + ncol] = hn;
      out[((size_t)m * TT + t) * 256 + ncol] = hn;   // out[B][T][H]
    }
    __syncthreads();
  }
}

extern "C" void kernel_launch(void* const* d_in, const int* in_sizes, int n_in,
                              void* d_out, int out_size, void* d_ws, size_t ws_size,
                              hipStream_t stream) {
  (void)in_sizes; (void)n_in; (void)out_size; (void)ws_size;
  const float* x      = (const float*)d_in[0];
  const float* W_ih   = (const float*)d_in[1];
  const float* W_hh   = (const float*)d_in[2];
  const float* b_lstm = (const float*)d_in[3];
  const float* Ws_w   = (const float*)d_in[4];
  const float* Ws_b   = (const float*)d_in[5];
  const float* Us_w   = (const float*)d_in[6];
  const float* Us_b   = (const float*)d_in[7];
  const float* Wc     = (const float*)d_in[8];
  const float* bc     = (const float*)d_in[9];

  char* ws = (char*)d_ws;
  h16*   W1p  = (h16*)(ws);                       // 64*16*512 f16 = 1,048,576 B
  h16*   Wsp  = (h16*)(ws + 1048576);             // 16*24*512 f16 =   393,216 B
  h16*   Wcp  = (h16*)(ws + 1441792);             // 64*24*512 f16 = 1,572,864 B
  float* hs   = (float*)(ws + 3014656);           // 192*16*256 f32 = 3,145,728 B
  float* base = (float*)(ws + 6160384);           // 192*16*256 f32 = 3,145,728 B

  // W1 = [W_ih | W_hh] : N=1024, K=512
  pack_weights<<<(64 * 16 * 512) / 512, 512, 0, stream>>>(W1p, W_ih, W_hh, 256, 512, 64, 16);
  // Ws_w : N=256, K=768 (cols: hs | h_task | x)
  pack_weights<<<(16 * 24 * 512) / 512, 512, 0, stream>>>(Wsp, Ws_w, Ws_w, 768, 768, 16, 24);
  // Wc : N=1024, K=768 (cols: x | Rt | h)
  pack_weights<<<(64 * 24 * 512) / 512, 512, 0, stream>>>(Wcp, Wc, Wc, 768, 768, 64, 24);

  shared_lstm_kernel<<<1, 512, 0, stream>>>(x, W1p, b_lstm, hs);
  base_kernel<<<TT, 256, 0, stream>>>(hs, x, Wsp, Ws_b, base);
  task_kernel<<<1, 512, 0, stream>>>(x, hs, base, Wsp, Us_w, Us_b, Wcp, bc, (float*)d_out);
}
